// MultiHeadAttention_42451456754056
// MI455X (gfx1250) — matrix operand order repaired and checked by reading, not compiled, not run on hardware
//
#include <hip/hip_runtime.h>

// ---------------------------------------------------------------------------
// MHA forward for MI455X (gfx1250): bf16 WMMA 16x16x32 + TDM (tensor_load_to_lds)
// double-buffered GEMM staging + permlane16 softmax reductions.
// B=2, S=2048, D=1024, H=16, DH=64 (hardcoded; all tiles divide exactly).
// ---------------------------------------------------------------------------

typedef __bf16 bf16;
typedef __attribute__((ext_vector_type(16))) __bf16 v16bf;
typedef __attribute__((ext_vector_type(8)))  float  v8f;
typedef __attribute__((ext_vector_type(4)))  unsigned int u32x4;
typedef __attribute__((ext_vector_type(8)))  int  i32x8;
typedef __attribute__((ext_vector_type(4)))  int  i32x4;

union Frag16 { v16bf v; uint4 q[2]; bf16 h[16]; };

__device__ __forceinline__ v8f wmma_bf16(v16bf a, v16bf b, v8f c) {
  return __builtin_amdgcn_wmma_f32_16x16x32_bf16(false, a, false, b, (short)0, c,
                                                 false, false);
}

constexpr int BB = 2, S = 2048, D = 1024, H = 16, DH = 64;
constexpr int Mdim = BB * S;   // 4096
constexpr int Kdim = D;        // 1024
constexpr int Ndim = D;        // 1024

// ---------------------------------------------------------------------------
// TDM: issue a 2-D bf16 tile load (tile_dim0=32 elems = 64B rows) into LDS with
// hardware padding of 4 DWORDs per 16 DWORDs -> 80B LDS row stride (LDA=40).
// D# packing per CDNA5 ISA 8.3/8.4. Tracked by TENSORcnt.
// ---------------------------------------------------------------------------
__device__ __forceinline__ void tdm_load_tile32(unsigned lds_off, const bf16* gaddr,
                                                unsigned rows) {
  const unsigned long long ga = (unsigned long long)(uintptr_t)gaddr;
  u32x4 g0;
  g0[0] = 1u;                                   // count=1 (valid user descriptor)
  g0[1] = lds_off;                              // lds_addr (bytes)
  g0[2] = (unsigned)(ga & 0xffffffffu);         // global_addr[31:0]
  g0[3] = (unsigned)((ga >> 32) & 0x1ffffffu)   // global_addr[56:32]
          | (2u << 30);                         // type=2 ("image")
  const unsigned TD = 1u << 20;                 // generous tensor dims (no clipping)
  i32x8 g1;
  g1[0] = (int)((1u << 16)                      // data_size=1 -> 2 bytes
                | (1u << 20)                    // pad_enable
                | (3u << 22)                    // pad_interval=3 -> every 16 DWORDs
                | (3u << 25));                  // pad_amount=3 -> 4 DWORDs (16B)
  g1[1] = (int)((TD & 0xffffu) << 16);          // [63:48] tensor_dim0[15:0]
  g1[2] = (int)(((TD >> 16) & 0xffffu)          // [79:64] tensor_dim0[31:16]
                | ((TD & 0xffffu) << 16));      // [95:80] tensor_dim1[15:0]... (low 16)
  g1[3] = (int)(((TD >> 16) & 0xffffu)          // [111:96] tensor_dim1[31:16]
                | (32u << 16));                 // [127:112] tile_dim0 = 32 elems
  g1[4] = (int)(rows & 0xffffu);                // [143:128] tile_dim1 = rows; tile_dim2=0
  g1[5] = (int)Kdim;                            // tensor_dim0_stride[31:0] = 1024 elems
  g1[6] = 0;                                    // stride hi / dim1_stride lo
  g1[7] = 0;
  const i32x4 zz = {0, 0, 0, 0};
#if __has_include(<hip/amd_detail/amd_gfx1250_TDM.h>)
  const i32x8 z8 = {0, 0, 0, 0, 0, 0, 0, 0};
  __builtin_amdgcn_tensor_load_to_lds(g0, g1, zz, zz, z8, 0);   // therock 6-arg form
#else
  __builtin_amdgcn_tensor_load_to_lds(g0, g1, zz, zz, 0);       // ROCm 7.2 5-arg form
#endif
}

// ---------------------------------------------------------------------------
// In-row-of-16 xor shuffle on the VALU pipe (co-executes with WMMA/XDL).
// ---------------------------------------------------------------------------
__device__ __forceinline__ float row_xor(float v, int mask, unsigned s1, unsigned s2) {
#if __has_builtin(__builtin_amdgcn_permlane16)
  (void)mask;
  unsigned u = __float_as_uint(v);
  u = __builtin_amdgcn_permlane16(u, u, s1, s2, false, false);
  return __uint_as_float(u);
#else
  (void)s1; (void)s2;
  return __shfl_xor(v, mask, 32);
#endif
}
__device__ __forceinline__ float red_max16(float v) {
  v = fmaxf(v, row_xor(v, 1, 0x67452301u, 0xEFCDAB89u));
  v = fmaxf(v, row_xor(v, 2, 0x54761032u, 0xDCFE98BAu));
  v = fmaxf(v, row_xor(v, 4, 0x32107654u, 0xBA98FEDCu));
  v = fmaxf(v, row_xor(v, 8, 0xFEDCBA98u, 0x76543210u));
  return v;
}
__device__ __forceinline__ float red_sum16(float v) {
  v += row_xor(v, 1, 0x67452301u, 0xEFCDAB89u);
  v += row_xor(v, 2, 0x54761032u, 0xDCFE98BAu);
  v += row_xor(v, 4, 0x32107654u, 0xBA98FEDCu);
  v += row_xor(v, 8, 0xFEDCBA98u, 0x76543210u);
  return v;
}

// ---------------------------------------------------------------------------
// f32 -> bf16 convert (8 elems/thread).  Grid covers n exactly.
// ---------------------------------------------------------------------------
__global__ __launch_bounds__(256) void cvt_bf16_k(const float* __restrict__ in,
                                                  bf16* __restrict__ out) {
  const size_t i = ((size_t)blockIdx.x * 256 + threadIdx.x) * 8;
  float4 a = ((const float4*)(in + i))[0];
  float4 b = ((const float4*)(in + i))[1];
  union { bf16 h[8]; uint4 q; } pk;
  pk.h[0] = (bf16)a.x; pk.h[1] = (bf16)a.y; pk.h[2] = (bf16)a.z; pk.h[3] = (bf16)a.w;
  pk.h[4] = (bf16)b.x; pk.h[5] = (bf16)b.y; pk.h[6] = (bf16)b.z; pk.h[7] = (bf16)b.w;
  *(uint4*)(out + i) = pk.q;
}

// ---------------------------------------------------------------------------
// Weight transpose + convert:  Wt[n][k] = (bf16) W[k][n]
// ---------------------------------------------------------------------------
__global__ __launch_bounds__(256) void transpose_w(const float* __restrict__ W,
                                                   bf16* __restrict__ Wt) {
  __shared__ float t[16][17];
  const int n0 = blockIdx.x * 16, k0 = blockIdx.y * 16;
  const int tx = threadIdx.x, ty = threadIdx.y;
  t[ty][tx] = W[(size_t)(k0 + ty) * Ndim + (n0 + tx)];
  __syncthreads();
  Wt[(size_t)(n0 + ty) * Kdim + (k0 + tx)] = (bf16)t[tx][ty];
}

// ---------------------------------------------------------------------------
// TDM-staged GEMM: C[m][n] = sum_k A[m][k] * Bt[n][k], A & Bt bf16 row-major.
// Block 256 thr (8 waves), tile 128Mx64N, K-step 32, double-buffered LDS fed by
// tensor_load_to_lds; wave0 issues next-stage loads, waits TENSORcnt<=2.
// MODE 0: bf16 out [B,H,S,DH] | MODE 1: bf16 out [B,H,DH,S] | MODE 2: f32 [M,N]
// ---------------------------------------------------------------------------
template <int MODE>
__global__ __launch_bounds__(256) void gemm_tdm(const bf16* __restrict__ A,
                                                const bf16* __restrict__ Bt,
                                                void* __restrict__ Cout) {
  constexpr int LDA = 40;  // 80B padded row stride produced by TDM padding
  __shared__ __align__(16) bf16 As[2][128 * LDA];
  __shared__ __align__(16) bf16 Bs[2][64 * LDA];

  const int m0 = blockIdx.x * 128;
  const int n0 = blockIdx.y * 64;
  const int tid = threadIdx.x;
  const int wv = tid >> 5, lane = tid & 31;
  const int half = lane >> 4, ln = lane & 15;
  const int waveM = (wv >> 1) << 5;  // 4 waves along M
  const int waveN = (wv & 1) << 5;   // 2 waves along N

  const unsigned ldsA[2] = {(unsigned)(uintptr_t)&As[0][0], (unsigned)(uintptr_t)&As[1][0]};
  const unsigned ldsB[2] = {(unsigned)(uintptr_t)&Bs[0][0], (unsigned)(uintptr_t)&Bs[1][0]};
  const bf16* Ab = A + (size_t)m0 * Kdim;
  const bf16* Bb = Bt + (size_t)n0 * Kdim;

  if (wv == 0) {  // prologue: stage 0 into buffer 0
    tdm_load_tile32(ldsA[0], Ab, 128);
    tdm_load_tile32(ldsB[0], Bb, 64);
  }

  v8f acc[2][2] = {};

  for (int k0 = 0; k0 < Kdim; k0 += 32) {
    const int buf = (k0 >> 5) & 1;
    if (wv == 0) {
      if (k0 + 32 < Kdim) {  // issue next stage into the other buffer
        tdm_load_tile32(ldsA[buf ^ 1], Ab + k0 + 32, 128);
        tdm_load_tile32(ldsB[buf ^ 1], Bb + k0 + 32, 64);
        __builtin_amdgcn_s_wait_tensorcnt(2);  // retire current stage's 2 loads
      } else {
        __builtin_amdgcn_s_wait_tensorcnt(0);
      }
    }
    __syncthreads();

    Frag16 af[2], bfr[2];
#pragma unroll
    for (int mi = 0; mi < 2; ++mi) {
      const bf16* ar = &As[buf][(waveM + mi * 16 + ln) * LDA];
      af[mi].q[0] = *(const uint4*)(ar + half * 8);       // K = half*8 + 0..7
      af[mi].q[1] = *(const uint4*)(ar + 16 + half * 8);  // K = 16 + half*8 + 0..7
    }
#pragma unroll
    for (int ni = 0; ni < 2; ++ni) {
      const bf16* br = &Bs[buf][(waveN + ni * 16 + ln) * LDA];
      bfr[ni].q[0] = *(const uint4*)(br + half * 8);      // K = half*8 + 0..7
      bfr[ni].q[1] = *(const uint4*)(br + 16 + half * 8); // K = 16 + half*8 + 0..7
    }
#pragma unroll
    for (int mi = 0; mi < 2; ++mi)
#pragma unroll
      for (int ni = 0; ni < 2; ++ni)
        acc[mi][ni] = wmma_bf16(af[mi].v, bfr[ni].v, acc[mi][ni]);

    __syncthreads();
  }

  // Epilogue: C VGPR r -> row (r + 8*half), col = ln within each 16x16 tile.
#pragma unroll
  for (int mi = 0; mi < 2; ++mi)
#pragma unroll
    for (int ni = 0; ni < 2; ++ni) {
      const int gn = n0 + waveN + ni * 16 + ln;
#pragma unroll
      for (int r = 0; r < 8; ++r) {
        const int gm = m0 + waveM + mi * 16 + r + half * 8;
        const float val = acc[mi][ni][r];
        if (MODE == 2) {
          ((float*)Cout)[(size_t)gm * Ndim + gn] = val;
        } else {
          const int b = gm >> 11, s = gm & 2047;
          const int hh = gn >> 6, dh = gn & 63;
          size_t idx;
          if (MODE == 0)
            idx = (((size_t)(b * H + hh)) * S + s) * DH + dh;  // [B,H,S,DH]
          else
            idx = (((size_t)(b * H + hh)) * DH + dh) * S + s;  // [B,H,DH,S]
          ((bf16*)Cout)[idx] = (bf16)val;
        }
      }
    }
}

// ---------------------------------------------------------------------------
// Flash attention: 1 wave = 16 query rows; 32-key tiles; 4 score WMMAs +
// 4 PV WMMAs per tile; online softmax state per (VGPR-row r, lane-half).
// ---------------------------------------------------------------------------
__global__ __launch_bounds__(256) void attn_kernel(const bf16* __restrict__ Qh,
                                                   const bf16* __restrict__ Kh,
                                                   const bf16* __restrict__ Vt,
                                                   const int* __restrict__ vlen,
                                                   bf16* __restrict__ Ob) {
  __shared__ __align__(16) bf16 Plds[8][16 * 32];  // per-wave P re-layout scratch

  const int tid = threadIdx.x;
  const int wv = tid >> 5, lane = tid & 31;
  const int half = lane >> 4, ln = lane & 15;

  const int bh = blockIdx.x >> 4;    // b*H + h
  const int qblk = blockIdx.x & 15;  // 16 q-chunks of 128 rows
  const int b = bh >> 4, h = bh & 15;
  const int qbase = qblk * 128 + wv * 16;

  const bf16* Qp = Qh + (size_t)bh * S * DH;  // [S][DH]
  const bf16* Kp = Kh + (size_t)bh * S * DH;  // [S][DH]
  const bf16* Vp = Vt + (size_t)bh * DH * S;  // [DH][S]
  const int valid = vlen[b];

  Frag16 aq0, aq1;
  {
    const bf16* qr = Qp + (size_t)(qbase + ln) * DH;
    aq0.q[0] = *(const uint4*)(qr + half * 8);
    aq0.q[1] = *(const uint4*)(qr + 16 + half * 8);
    aq1.q[0] = *(const uint4*)(qr + 32 + half * 8);
    aq1.q[1] = *(const uint4*)(qr + 48 + half * 8);
  }

  float mrow[8], lrow[8];
  v8f o[4] = {};
#pragma unroll
  for (int r = 0; r < 8; ++r) { mrow[r] = -3.0e38f; lrow[r] = 0.0f; }

  for (int kb = 0; kb < S; kb += 32) {
    v8f sc[2];
#pragma unroll
    for (int cb = 0; cb < 2; ++cb) {
      const int krow = kb + cb * 16 + ln;
      const bf16* kr = Kp + (size_t)krow * DH;
      Frag16 b0, b1;
      b0.q[0] = *(const uint4*)(kr + half * 8);
      b0.q[1] = *(const uint4*)(kr + 16 + half * 8);
      b1.q[0] = *(const uint4*)(kr + 32 + half * 8);
      b1.q[1] = *(const uint4*)(kr + 48 + half * 8);
      v8f s = {};
      s = wmma_bf16(aq0.v, b0.v, s);
      s = wmma_bf16(aq1.v, b1.v, s);
      const bool oob = (krow >= valid);  // key-side padding mask
#pragma unroll
      for (int r = 0; r < 8; ++r) sc[cb][r] = oob ? -1.0e6f : s[r] * 0.125f;
      if (kb + 32 < S)
        __builtin_prefetch(Kp + (size_t)(krow + 32) * DH, 0, 1);
    }

    float aexp[8];
#pragma unroll
    for (int r = 0; r < 8; ++r) {
      const float tm = red_max16(fmaxf(sc[0][r], sc[1][r]));
      const float mn = fmaxf(mrow[r], tm);
      aexp[r] = __expf(mrow[r] - mn);
      mrow[r] = mn;
    }
#pragma unroll
    for (int r = 0; r < 8; ++r) {
      const float p0 = __expf(sc[0][r] - mrow[r]);
      const float p1 = __expf(sc[1][r] - mrow[r]);
      sc[0][r] = p0;
      sc[1][r] = p1;
      lrow[r] = lrow[r] * aexp[r] + red_sum16(p0 + p1);
    }
#pragma unroll
    for (int t = 0; t < 4; ++t)
#pragma unroll
      for (int r = 0; r < 8; ++r) o[t][r] *= aexp[r];

    // P (C-layout f32) -> LDS bf16 -> A-layout fragment (per-wave scratch).
    bf16* pb = &Plds[wv][0];
#pragma unroll
    for (int cb = 0; cb < 2; ++cb)
#pragma unroll
      for (int r = 0; r < 8; ++r)
        pb[(r + half * 8) * 32 + cb * 16 + ln] = (bf16)sc[cb][r];
    asm volatile("s_wait_dscnt 0" ::: "memory");  // DS in-order per wave; be explicit
    Frag16 apf;
    {
      const bf16* pr = pb + ln * 32;
      apf.q[0] = *(const uint4*)(pr + half * 8);
      apf.q[1] = *(const uint4*)(pr + 16 + half * 8);
    }

#pragma unroll
    for (int t = 0; t < 4; ++t) {
      const bf16* vr = Vp + (size_t)(t * 16 + ln) * S + kb;
      Frag16 bv;
      bv.q[0] = *(const uint4*)(vr + half * 8);
      bv.q[1] = *(const uint4*)(vr + 16 + half * 8);
      o[t] = wmma_bf16(apf.v, bv.v, o[t]);
    }
  }

#pragma unroll
  for (int t = 0; t < 4; ++t)
#pragma unroll
    for (int r = 0; r < 8; ++r) {
      const int srow = qbase + r + half * 8;
      const int col = h * DH + t * 16 + ln;
      Ob[((size_t)b * S + srow) * D + col] = (bf16)(o[t][r] / lrow[r]);
    }
}

// ---------------------------------------------------------------------------
// Host-side launcher
// ---------------------------------------------------------------------------
extern "C" void kernel_launch(void* const* d_in, const int* in_sizes, int n_in,
                              void* d_out, int out_size, void* d_ws, size_t ws_size,
                              hipStream_t stream) {
  (void)in_sizes; (void)n_in; (void)out_size; (void)ws_size;
  const float* queries = (const float*)d_in[0];
  const float* keys    = (const float*)d_in[1];
  const float* values  = (const float*)d_in[2];
  const float* Wq = (const float*)d_in[3];
  const float* Wk = (const float*)d_in[4];
  const float* Wv = (const float*)d_in[5];
  const float* Wo = (const float*)d_in[6];
  const int*   vl = (const int*)d_in[7];

  char* ws = (char*)d_ws;
  bf16* Wqt = (bf16*)(ws + ((size_t)0 << 20));   // 2 MB each, bf16 [N][K]
  bf16* Wkt = (bf16*)(ws + ((size_t)2 << 20));
  bf16* Wvt = (bf16*)(ws + ((size_t)4 << 20));
  bf16* Wot = (bf16*)(ws + ((size_t)6 << 20));
  bf16* Xq  = (bf16*)(ws + ((size_t)8 << 20));   // bf16 activations, 8 MB each
  bf16* Xk  = (bf16*)(ws + ((size_t)16 << 20));
  bf16* Xv  = (bf16*)(ws + ((size_t)24 << 20));
  bf16* Qh  = (bf16*)(ws + ((size_t)32 << 20));  // [B,H,S,DH]
  bf16* Kh  = (bf16*)(ws + ((size_t)40 << 20));  // [B,H,S,DH]
  bf16* Vt  = (bf16*)(ws + ((size_t)48 << 20));  // [B,H,DH,S]
  bf16* Ab  = (bf16*)(ws + ((size_t)56 << 20));  // attention out, bf16 [B,S,D]

  const int nAct = Mdim * Kdim;  // 4194304
  cvt_bf16_k<<<nAct / (256 * 8), 256, 0, stream>>>(queries, Xq);
  cvt_bf16_k<<<nAct / (256 * 8), 256, 0, stream>>>(keys, Xk);
  cvt_bf16_k<<<nAct / (256 * 8), 256, 0, stream>>>(values, Xv);

  dim3 tb(16, 16), tg(Ndim / 16, Kdim / 16);
  transpose_w<<<tg, tb, 0, stream>>>(Wq, Wqt);
  transpose_w<<<tg, tb, 0, stream>>>(Wk, Wkt);
  transpose_w<<<tg, tb, 0, stream>>>(Wv, Wvt);
  transpose_w<<<tg, tb, 0, stream>>>(Wo, Wot);

  dim3 gg(Mdim / 128, Ndim / 64);
  gemm_tdm<0><<<gg, 256, 0, stream>>>(Xq, Wqt, (void*)Qh);
  gemm_tdm<0><<<gg, 256, 0, stream>>>(Xk, Wkt, (void*)Kh);
  gemm_tdm<1><<<gg, 256, 0, stream>>>(Xv, Wvt, (void*)Vt);

  attn_kernel<<<BB * H * (S / 128), 256, 0, stream>>>(Qh, Kh, Vt, vl, Ab);

  gemm_tdm<2><<<gg, 256, 0, stream>>>(Ab, Wot, d_out);
}
